// PVConv_31456340476522
// MI455X (gfx1250) — matrix operand order repaired and checked by reading, not compiled, not run on hardware
//
#include <hip/hip_runtime.h>
#include <hip/hip_bf16.h>

#define R3 32768
#define NPTS 32768
#define NB 2
#define LSLOPE 0.1f
#define BNEPS 1e-4f

typedef _Float16 f16x16 __attribute__((ext_vector_type(16)));
typedef _Float16 f16x8  __attribute__((ext_vector_type(8)));
typedef float    f32x8  __attribute__((ext_vector_type(8)));
typedef float    f32x4  __attribute__((ext_vector_type(4)));

// ---------------- utility kernels ----------------

__global__ void zero_kernel(float* __restrict__ p, int count) {
    int t = blockIdx.x * 256 + threadIdx.x;
    if (t < count) p[t] = 0.0f;
}

// repack conv weights f32 [Cout][Cin][3][3][3] -> f16 [Cout][tap*Cin+ci]
__global__ void prep_w_kernel(const float* __restrict__ w, _Float16* __restrict__ wh,
                              int Cin, int total) {
    int t = blockIdx.x * 256 + threadIdx.x;
    if (t >= total) return;
    int per = Cin * 27;
    int co  = t / per;
    int r   = t % per;
    int ci  = r / 27;
    int tap = r % 27;
    wh[(size_t)co * per + tap * Cin + ci] = (_Float16)w[t];
}

// per-(b,dim) mean of coords [B,3,N]
__global__ void mean_kernel(const float* __restrict__ coords, float* __restrict__ cmean) {
    __shared__ float s[256];
    int bd = blockIdx.x;                       // 0..5
    const float* p = coords + (size_t)bd * NPTS;
    float acc = 0.f;
    for (int i = threadIdx.x; i < NPTS; i += 256) acc += p[i];
    s[threadIdx.x] = acc;
    __syncthreads();
    for (int off = 128; off > 0; off >>= 1) {
        if (threadIdx.x < off) s[threadIdx.x] += s[threadIdx.x + off];
        __syncthreads();
    }
    if (threadIdx.x == 0) cmean[bd] = s[0] / (float)NPTS;
}

// per-batch scale = 2 * max ||centered||
__global__ void scale_kernel(const float* __restrict__ coords,
                             const float* __restrict__ cmean, float* __restrict__ scale) {
    __shared__ float s[256];
    int b = blockIdx.x;
    float mx = cmean[b*3+0], my = cmean[b*3+1], mz = cmean[b*3+2];
    const float* px = coords + (size_t)(b*3) * NPTS;
    float best = 0.f;
    for (int i = threadIdx.x; i < NPTS; i += 256) {
        float cx = px[i] - mx;
        float cy = px[NPTS + i] - my;
        float cz = px[2*NPTS + i] - mz;
        float n2 = cx*cx + cy*cy + cz*cz;
        best = fmaxf(best, n2);
    }
    s[threadIdx.x] = best;
    __syncthreads();
    for (int off = 128; off > 0; off >>= 1) {
        if (threadIdx.x < off) s[threadIdx.x] = fmaxf(s[threadIdx.x], s[threadIdx.x + off]);
        __syncthreads();
    }
    if (threadIdx.x == 0) scale[b] = 2.0f * sqrtf(s[0]);
}

// compute norm_coords, scatter-add features into gridSum/counts
__global__ void voxelize_kernel(const float* __restrict__ feats, const float* __restrict__ coords,
                                const float* __restrict__ cmean, const float* __restrict__ scale,
                                float* __restrict__ ncoords, float* __restrict__ gsum,
                                float* __restrict__ cnt) {
    int t = blockIdx.x * 256 + threadIdx.x;      // < NB*NPTS
    int b = t >> 15, n = t & (NPTS - 1);
    float inv = 1.0f / scale[b];
    int vox[3];
    #pragma unroll
    for (int d = 0; d < 3; ++d) {
        float c = coords[((size_t)(b*3 + d)) * NPTS + n];
        float v = (c - cmean[b*3 + d]) * inv + 0.5f;
        v = v * 32.0f;
        v = fminf(fmaxf(v, 0.0f), 31.0f);
        ncoords[((size_t)b * NPTS + n) * 3 + d] = v;
        vox[d] = (int)rintf(v);
    }
    int flat = (vox[0] * 32 + vox[1]) * 32 + vox[2];
    size_t base = (size_t)b * R3 + flat;
    atomicAdd(cnt + base, 1.0f);
    float* gs = gsum + base * 64;
    const float* f = feats + (size_t)b * 64 * NPTS + n;
    #pragma unroll 4
    for (int c = 0; c < 64; ++c) atomicAdd(gs + c, f[(size_t)c * NPTS]);
}

// gridSum/count -> f16 channel-last grid
__global__ void finalize_kernel(const float* __restrict__ gsum, const float* __restrict__ cnt,
                                _Float16* __restrict__ gridh) {
    int t = blockIdx.x * 256 + threadIdx.x;      // < NB*R3*64
    int i = t >> 6;
    float c = fmaxf(cnt[i], 1.0f);
    gridh[t] = (_Float16)(gsum[t] / c);
}

// ---------------- implicit-GEMM conv3d via WMMA ----------------
// grid: f16 [b][vox][CIN] channel-last; wh: f16 [128][27*CIN]; yout: f32 [b][vox][128]
template <int CIN>
__global__ __launch_bounds__(256)
void conv3d_wmma_kernel(const _Float16* __restrict__ grid, const _Float16* __restrict__ wh,
                        const float* __restrict__ bias, float* __restrict__ yout) {
    constexpr int KTOT = 27 * CIN;
    const int lane = threadIdx.x & 31;
    const int wid  = threadIdx.x >> 5;
    const int gw   = blockIdx.x * 8 + wid;       // 0..32767
    const int voxTile = gw & 2047;
    const int coTile  = (gw >> 11) & 7;
    const int b       = gw >> 14;

    const int vx = voxTile >> 6;                 // 0..31
    const int vy = (voxTile >> 1) & 31;          // 0..31
    const int z0 = (voxTile & 1) << 4;           // 0 or 16
    const int mn  = lane & 15;                   // A-row m / B,C-column n
    const int khl = lane >> 4;                   // K half / C row-group

    f32x8 acc;
    #pragma unroll
    for (int v = 0; v < 8; ++v) acc[v] = bias[coTile * 16 + khl * 8 + v];

    const _Float16* wbase = wh + (size_t)(coTile * 16 + mn) * KTOT;
    const size_t gbatch = (size_t)b * R3;

    for (int tap = 0; tap < 27; ++tap) {
        const int td = tap / 9, th = (tap / 3) % 3, tw = tap % 3;
        const int xx = vx + td - 1;
        const int yy = vy + th - 1;
        const int zz = z0 + mn + tw - 1;
        const bool valid = ((unsigned)xx < 32u) & ((unsigned)yy < 32u) & ((unsigned)zz < 32u);
        const _Float16* gb = grid + (gbatch + (size_t)(((xx * 32 + yy) * 32) + zz)) * CIN + khl * 16;
        const _Float16* wb = wbase + tap * CIN;
        #pragma unroll
        for (int ch = 0; ch < CIN / 32; ++ch) {
            union { f16x16 v; f16x8 h[2]; } A, Bm;
            A.h[0] = *(const f16x8*)(wb + ch * 32 + khl * 16);
            A.h[1] = *(const f16x8*)(wb + ch * 32 + khl * 16 + 8);
            f16x8 z8 = {};
            Bm.h[0] = z8;
            Bm.h[1] = z8;
            if (valid) {
                Bm.h[0] = *(const f16x8*)(gb + ch * 32);
                Bm.h[1] = *(const f16x8*)(gb + ch * 32 + 8);
            }
            acc = __builtin_amdgcn_wmma_f32_16x16x32_f16(false, A.v, false, Bm.v,
                                                         (short)0, acc, false, false);
        }
    }

    float* yp = yout + (gbatch + (size_t)(((vx * 32 + vy) * 32) + z0 + mn)) * 128
                     + coTile * 16 + khl * 8;
    f32x4 lo = {acc[0], acc[1], acc[2], acc[3]};
    f32x4 hi = {acc[4], acc[5], acc[6], acc[7]};
    *(f32x4*)yp       = lo;
    *(f32x4*)(yp + 4) = hi;
}

// ---------------- BN statistics ----------------

// y layout [i][128], i over NB*R3 -> mean/rstd per channel
__global__ void stats_y_kernel(const float* __restrict__ y,
                               float* __restrict__ mean, float* __restrict__ rstd) {
    __shared__ float ssum[256], ssq[256];
    int c = blockIdx.x;
    float s = 0.f, q = 0.f;
    for (int i = threadIdx.x; i < NB * R3; i += 256) {
        float v = y[(size_t)i * 128 + c];
        s += v; q += v * v;
    }
    ssum[threadIdx.x] = s; ssq[threadIdx.x] = q;
    __syncthreads();
    for (int off = 128; off > 0; off >>= 1) {
        if (threadIdx.x < off) {
            ssum[threadIdx.x] += ssum[threadIdx.x + off];
            ssq[threadIdx.x]  += ssq[threadIdx.x + off];
        }
        __syncthreads();
    }
    if (threadIdx.x == 0) {
        float m = ssum[0] / (float)(NB * R3);
        float var = ssq[0] / (float)(NB * R3) - m * m;
        mean[c] = m;
        rstd[c] = rsqrtf(var + BNEPS);
    }
}

// p layout [b][c][n] -> mean/rstd per channel over (b,n)
__global__ void stats_p_kernel(const float* __restrict__ p,
                               float* __restrict__ mean, float* __restrict__ rstd) {
    __shared__ float ssum[256], ssq[256];
    int c = blockIdx.x;
    float s = 0.f, q = 0.f;
    for (int i = threadIdx.x; i < NB * NPTS; i += 256) {
        size_t addr = (i < NPTS) ? ((size_t)c * NPTS + i)
                                 : ((size_t)(128 + c) * NPTS + (i - NPTS));
        float v = p[addr];
        s += v; q += v * v;
    }
    ssum[threadIdx.x] = s; ssq[threadIdx.x] = q;
    __syncthreads();
    for (int off = 128; off > 0; off >>= 1) {
        if (threadIdx.x < off) {
            ssum[threadIdx.x] += ssum[threadIdx.x + off];
            ssq[threadIdx.x]  += ssq[threadIdx.x + off];
        }
        __syncthreads();
    }
    if (threadIdx.x == 0) {
        float m = ssum[0] / (float)(NB * NPTS);
        float var = ssq[0] / (float)(NB * NPTS) - m * m;
        mean[c] = m;
        rstd[c] = rsqrtf(var + BNEPS);
    }
}

// BN + LeakyReLU, write f16 (channel-last)
__global__ void bn_apply_h_kernel(const float* __restrict__ y,
                                  const float* __restrict__ mean, const float* __restrict__ rstd,
                                  const float* __restrict__ g, const float* __restrict__ be,
                                  _Float16* __restrict__ out) {
    int t = blockIdx.x * 256 + threadIdx.x;      // < NB*R3*128
    int c = t & 127;
    float v = (y[t] - mean[c]) * rstd[c] * g[c] + be[c];
    v = v > 0.f ? v : LSLOPE * v;
    out[t] = (_Float16)v;
}

// BN + LeakyReLU, in-place f32
__global__ void bn_apply_f_kernel(float* __restrict__ y,
                                  const float* __restrict__ mean, const float* __restrict__ rstd,
                                  const float* __restrict__ g, const float* __restrict__ be) {
    int t = blockIdx.x * 256 + threadIdx.x;
    int c = t & 127;
    float v = (y[t] - mean[c]) * rstd[c] * g[c] + be[c];
    y[t] = v > 0.f ? v : LSLOPE * v;
}

// ---------------- point branch and devoxelize ----------------

__global__ void pmm_kernel(const float* __restrict__ feats, const float* __restrict__ wp,
                           const float* __restrict__ bp, float* __restrict__ p) {
    int t = blockIdx.x * 256 + threadIdx.x;      // < NB*128*NPTS
    int n = t & (NPTS - 1);
    int c = (t >> 15) & 127;
    int b = t >> 22;
    const float* f = feats + (size_t)b * 64 * NPTS + n;
    const float* w = wp + c * 64;
    float acc = bp[c];
    #pragma unroll
    for (int ci = 0; ci < 64; ++ci) acc += w[ci] * f[(size_t)ci * NPTS];
    p[t] = acc;
}

// trilinear gather: v [b][vox][128] -> out [b][c][n]
__global__ void devox_kernel(const float* __restrict__ v, const float* __restrict__ nc,
                             float* __restrict__ out) {
    int t = blockIdx.x * 256 + threadIdx.x;      // < NB*NPTS*32
    int cq = t & 31;
    int n  = (t >> 5) & (NPTS - 1);
    int b  = t >> 20;
    const float* p = nc + ((size_t)b * NPTS + n) * 3;
    float cx = p[0], cy = p[1], cz = p[2];
    int lx = (int)floorf(cx), ly = (int)floorf(cy), lz = (int)floorf(cz);
    int hx = min(lx + 1, 31), hy = min(ly + 1, 31), hz = min(lz + 1, 31);
    float fx = cx - (float)lx, fy = cy - (float)ly, fz = cz - (float)lz;
    f32x4 acc = {};
    #pragma unroll
    for (int k = 0; k < 8; ++k) {
        int   ix = (k & 4) ? hx : lx; float wx = (k & 4) ? fx : 1.f - fx;
        int   iy = (k & 2) ? hy : ly; float wy = (k & 2) ? fy : 1.f - fy;
        int   iz = (k & 1) ? hz : lz; float wz = (k & 1) ? fz : 1.f - fz;
        float w = wx * wy * wz;
        const f32x4 g = *(const f32x4*)(v + ((size_t)b * R3 + (size_t)((ix * 32 + iy) * 32 + iz)) * 128 + cq * 4);
        acc += g * w;
    }
    size_t ob = ((size_t)(b * 128 + cq * 4)) * NPTS + n;
    out[ob]            = acc[0];
    out[ob + NPTS]     = acc[1];
    out[ob + 2 * NPTS] = acc[2];
    out[ob + 3 * NPTS] = acc[3];
}

// out += leaky(bn(p))   (both [b][c][n] flat)
__global__ void fuse_kernel(const float* __restrict__ p,
                            const float* __restrict__ mean, const float* __restrict__ rstd,
                            const float* __restrict__ g, const float* __restrict__ be,
                            float* __restrict__ out) {
    int t = blockIdx.x * 256 + threadIdx.x;      // < NB*128*NPTS
    int c = (t >> 15) & 127;
    float v = (p[t] - mean[c]) * rstd[c] * g[c] + be[c];
    v = v > 0.f ? v : LSLOPE * v;
    out[t] += v;
}

__global__ void copy_kernel(const float* __restrict__ src, float* __restrict__ dst, int count) {
    int t = blockIdx.x * 256 + threadIdx.x;
    if (t < count) dst[t] = src[t];
}

// ---------------- host side ----------------

extern "C" void kernel_launch(void* const* d_in, const int* in_sizes, int n_in,
                              void* d_out, int out_size, void* d_ws, size_t ws_size,
                              hipStream_t stream) {
    (void)in_sizes; (void)n_in; (void)out_size; (void)ws_size;
    const float* feats  = (const float*)d_in[0];
    const float* coords = (const float*)d_in[1];
    const float* w1  = (const float*)d_in[2];
    const float* b1  = (const float*)d_in[3];
    const float* g1  = (const float*)d_in[4];
    const float* be1 = (const float*)d_in[5];
    const float* w2  = (const float*)d_in[6];
    const float* b2  = (const float*)d_in[7];
    const float* g2  = (const float*)d_in[8];
    const float* be2 = (const float*)d_in[9];
    const float* wp  = (const float*)d_in[10];
    const float* bp  = (const float*)d_in[11];
    const float* gp  = (const float*)d_in[12];
    const float* bep = (const float*)d_in[13];
    float* out = (float*)d_out;

    char* ws = (char*)d_ws;
    auto align256 = [](size_t x) { return (x + 255) & ~(size_t)255; };
    size_t off = 0;
    float*     nc    = (float*)(ws + off);     off = align256(off + (size_t)NB * NPTS * 3 * 4);
    float*     cnt   = (float*)(ws + off);     off = align256(off + (size_t)NB * R3 * 4);
    float*     stats = (float*)(ws + off);     off = align256(off + 1024 * 4);
    _Float16*  wh1   = (_Float16*)(ws + off);  off = align256(off + (size_t)128 * 27 * 64 * 2);
    _Float16*  wh2   = (_Float16*)(ws + off);  off = align256(off + (size_t)128 * 27 * 128 * 2);
    float*     gsum  = (float*)(ws + off);     off = align256(off + (size_t)NB * R3 * 64 * 4);
    _Float16*  gridh = (_Float16*)(ws + off);  off = align256(off + (size_t)NB * R3 * 64 * 2);
    float*     ybuf  = (float*)(ws + off);     off = align256(off + (size_t)NB * R3 * 128 * 4);
    _Float16*  g2h   = (_Float16*)(ws + off);  off = align256(off + (size_t)NB * R3 * 128 * 2);
    float*     praw  = (float*)(ws + off);     off = align256(off + (size_t)NB * NPTS * 128 * 4);

    float* mean1 = stats;       float* rstd1 = stats + 128;
    float* mean2 = stats + 256; float* rstd2 = stats + 384;
    float* meanP = stats + 512; float* rstdP = stats + 640;
    float* cmean = stats + 768; float* scale = stats + 776;

    // 1) zero scatter accumulators
    zero_kernel<<<(NB * R3 * 64 + 255) / 256, 256, 0, stream>>>(gsum, NB * R3 * 64);
    zero_kernel<<<(NB * R3 + 255) / 256, 256, 0, stream>>>(cnt, NB * R3);

    // 2) repack conv weights to f16 [Cout][tap*Cin+ci]
    prep_w_kernel<<<(128 * 64 * 27 + 255) / 256, 256, 0, stream>>>(w1, wh1, 64, 128 * 64 * 27);
    prep_w_kernel<<<(128 * 128 * 27 + 255) / 256, 256, 0, stream>>>(w2, wh2, 128, 128 * 128 * 27);

    // 3) voxelization
    mean_kernel<<<NB * 3, 256, 0, stream>>>(coords, cmean);
    scale_kernel<<<NB, 256, 0, stream>>>(coords, cmean, scale);
    voxelize_kernel<<<(NB * NPTS + 255) / 256, 256, 0, stream>>>(feats, coords, cmean, scale,
                                                                 nc, gsum, cnt);
    finalize_kernel<<<(NB * R3 * 64 + 255) / 256, 256, 0, stream>>>(gsum, cnt, gridh);

    // 4) conv1 -> BN -> leaky -> f16
    conv3d_wmma_kernel<64><<<4096, 256, 0, stream>>>(gridh, wh1, b1, ybuf);
    stats_y_kernel<<<128, 256, 0, stream>>>(ybuf, mean1, rstd1);
    bn_apply_h_kernel<<<(NB * R3 * 128 + 255) / 256, 256, 0, stream>>>(ybuf, mean1, rstd1,
                                                                       g1, be1, g2h);

    // 5) conv2 -> BN -> leaky (f32 in place)
    conv3d_wmma_kernel<128><<<4096, 256, 0, stream>>>(g2h, wh2, b2, ybuf);
    stats_y_kernel<<<128, 256, 0, stream>>>(ybuf, mean2, rstd2);
    bn_apply_f_kernel<<<(NB * R3 * 128 + 255) / 256, 256, 0, stream>>>(ybuf, mean2, rstd2,
                                                                       g2, be2);

    // 6) point branch
    pmm_kernel<<<(NB * 128 * NPTS + 255) / 256, 256, 0, stream>>>(feats, wp, bp, praw);
    stats_p_kernel<<<128, 256, 0, stream>>>(praw, meanP, rstdP);

    // 7) devoxelize into output, then fuse point branch
    devox_kernel<<<(NB * NPTS * 32 + 255) / 256, 256, 0, stream>>>(ybuf, nc, out);
    fuse_kernel<<<(NB * 128 * NPTS + 255) / 256, 256, 0, stream>>>(praw, meanP, rstdP,
                                                                   gp, bep, out);

    // 8) coords pass-through
    copy_kernel<<<(NB * 3 * NPTS + 255) / 256, 256, 0, stream>>>(coords,
                                                                 out + (size_t)NB * 128 * NPTS,
                                                                 NB * 3 * NPTS);
}
